// CustomRNN_55336358643308
// MI455X (gfx1250) — compile-verified, matrix operand order checked by reading, not run on hardware
//
#include <hip/hip_runtime.h>
#include <hip/hip_bf16.h>
#include <math.h>

// MI455X / gfx1250: wave32, WMMA 16x16x32 bf16 with f32 accumulate.
typedef __attribute__((ext_vector_type(16))) __bf16 v16bf;
typedef __attribute__((ext_vector_type(8)))  float  v8f;
typedef __attribute__((ext_vector_type(4)))  int    v4i;

#define T_LEN 2048
#define D_DIM 512
#define U_DIM 512
#define B_SZ  128

// --- CDNA5 async global->LDS copy (ASYNCcnt) with portable fallback --------
// Builtin signature (from hipcc diagnostic): param0 = v4i in AS1 (global),
// param1 expected analogously in AS3 (LDS), then imm offset + imm cpol.
__device__ __forceinline__ void h_async_b128(const __bf16* g, __bf16* l) {
#if __has_builtin(__builtin_amdgcn_global_load_async_to_lds_b128)
  typedef v4i __attribute__((address_space(1)))* gas1;
  typedef v4i __attribute__((address_space(3)))* las3;
  __builtin_amdgcn_global_load_async_to_lds_b128((gas1)(void*)g, (las3)(void*)l, 0, 0);
#else
  *(int4*)l = *(const int4*)g;  // synchronous staging fallback
#endif
}
__device__ __forceinline__ void h_async_wait() {
#if __has_builtin(__builtin_amdgcn_s_wait_asynccnt)
  __builtin_amdgcn_s_wait_asynccnt(0);
#endif
}

// ---------------------------------------------------------------------------
// Kernel 0: transpose + downconvert weights to bf16 [N][K] so WMMA B-fragments
// (lane = one output column, 16 consecutive K) are contiguous 32-byte loads.
// ---------------------------------------------------------------------------
__global__ __launch_bounds__(256) void prep_weights_kernel(
    const float* __restrict__ W1, const float* __restrict__ W2,
    __bf16* __restrict__ W1t, __bf16* __restrict__ W2t) {
  int idx = blockIdx.x * blockDim.x + threadIdx.x;  // 0 .. 512*512-1
  int n = idx >> 9;
  int k = idx & 511;
  W1t[idx] = (__bf16)W1[k * U_DIM + n];
  W2t[idx] = (__bf16)W2[k * U_DIM + n];
}

// ---------------------------------------------------------------------------
// Kernel 1: H = tanh(X @ W1 + b1), X = [B*T, D] = [262144, 512].
// 64-row M-tile per block (4096 blocks), 16 waves; wave owns 2 N-tiles and
// 4 M-subtiles -> each B-fragment feeds 4 WMMAs (L2 traffic for W1t: ~2 GB).
// Output H is written TIME-MAJOR [T][B][U] so the sequential recurrence
// kernel sees one contiguous 16KB tile per step per workgroup.
// ---------------------------------------------------------------------------
__global__ __launch_bounds__(512) void input_gemm_tanh_kernel(
    const float* __restrict__ x, const __bf16* __restrict__ W1t,
    const float* __restrict__ b1, __bf16* __restrict__ H) {
  __shared__ __bf16 x_lds[64 * D_DIM];  // 64 KB

  const int tid  = threadIdx.x;
  const int lane = tid & 31;
  const int wv   = tid >> 5;        // 0..15
  const int nloc = lane & 15;
  const int kh   = lane >> 4;       // K-half (0/1)
  const size_t mbase = (size_t)blockIdx.x * 64;
  const int bidx  = (int)(mbase >> 11);   // batch index (64 | 2048)
  const int tbase = (int)(mbase & 2047);  // time base within sequence

  // Stage + convert the 64x512 input tile (coalesced fp32 reads).
  for (int i = tid; i < 64 * D_DIM; i += 512)
    x_lds[i] = (__bf16)x[(mbase + (size_t)(i >> 9)) * D_DIM + (i & 511)];
  __syncthreads();

  const int n0 = wv * 32 + nloc;    // first owned column
  const int n1 = n0 + 16;           // second owned column

  v8f zero = {};
  v8f acc[2][4];
  #pragma unroll
  for (int s = 0; s < 2; ++s)
    #pragma unroll
    for (int mt = 0; mt < 4; ++mt) acc[s][mt] = zero;

  #pragma unroll
  for (int kc = 0; kc < 16; ++kc) {
    const int k0 = kc * 32 + kh * 16;
    v16bf bf0 = *(const v16bf*)(W1t + (size_t)n0 * D_DIM + k0);
    v16bf bf1 = *(const v16bf*)(W1t + (size_t)n1 * D_DIM + k0);
    #pragma unroll
    for (int mt = 0; mt < 4; ++mt) {
      v16bf a = *(const v16bf*)(&x_lds[(mt * 16 + (lane & 15)) * D_DIM + k0]);
      acc[0][mt] = __builtin_amdgcn_wmma_f32_16x16x32_bf16(
          false, a, false, bf0, (short)0, acc[0][mt], false, false);
      acc[1][mt] = __builtin_amdgcn_wmma_f32_16x16x32_bf16(
          false, a, false, bf1, (short)0, acc[1][mt], false, false);
    }
  }

  const float bias0 = b1[n0], bias1 = b1[n1];
  #pragma unroll
  for (int mt = 0; mt < 4; ++mt) {
    #pragma unroll
    for (int r = 0; r < 8; ++r) {   // D layout: local M = r + 8*kh
      const int t = tbase + mt * 16 + r + 8 * kh;
      const size_t row = (size_t)t * B_SZ + bidx;   // time-major H
      H[row * U_DIM + n0] = (__bf16)tanhf(acc[0][mt][r] + bias0);
      H[row * U_DIM + n1] = (__bf16)tanhf(acc[1][mt][r] + bias1);
    }
  }
}

// ---------------------------------------------------------------------------
// Kernel 2: persistent-RNN recurrence fused with the final projection.
//   y_t = H_t + tanh(y_{t-1} @ W2 + b2);  out[b,t] = y_t . Wc + bc
// 8 blocks x 16 batch rows; 512 threads = 16 waves (4/SIMD -> ~512-VGPR
// budget), wave w owns N-tiles {2w, 2w+1}: Bfrag[2][16] = 256 VGPRs of W2
// stay register-resident for all 2048 steps; one shared A-load feeds 2 WMMAs.
// H tiles are DMA'd global->LDS asynchronously (double-buffered), overlapping
// the WMMA phase; state y lives in LDS as bf16.
// ---------------------------------------------------------------------------
__global__ __launch_bounds__(512) void recurrence_kernel(
    const __bf16* __restrict__ H, const __bf16* __restrict__ W2t,
    const float* __restrict__ b2, const float* __restrict__ Wc,
    const float* __restrict__ bc, float* __restrict__ out) {
  __shared__ __bf16 y_lds[16 * U_DIM];       // recurrent state, 16 KB
  __shared__ __bf16 h_tile[2][16 * U_DIM];   // double-buffered H, 32 KB
  __shared__ float accum[16];                // per-row dot(y, Wc)

  const int tid  = threadIdx.x;
  const int lane = tid & 31;
  const int wv   = tid >> 5;       // 0..15
  const int nloc = lane & 15;
  const int kh   = lane >> 4;
  const int n0   = wv * 32 + nloc;
  const int n1   = n0 + 16;
  const int rowbase = blockIdx.x * 16;       // batch rows owned by this block

  // One-time load of this wave's W2 slices; resident in VGPRs across all T.
  v16bf Bf0[16], Bf1[16];
  #pragma unroll
  for (int kc = 0; kc < 16; ++kc) {
    const int k0 = kc * 32 + kh * 16;
    Bf0[kc] = *(const v16bf*)(W2t + (size_t)n0 * U_DIM + k0);
    Bf1[kc] = *(const v16bf*)(W2t + (size_t)n1 * U_DIM + k0);
  }
  const float bias0 = b2[n0], bias1 = b2[n1];
  const float wc0 = Wc[n0],  wc1 = Wc[n1];
  const float bcv = bc[0];

  for (int i = tid; i < 16 * U_DIM; i += 512) y_lds[i] = (__bf16)0.0f;
  if (tid < 16) accum[tid] = 0.0f;

  // Preload H tile for t=0 (16KB contiguous; 2 x b128 per thread).
  {
    const __bf16* src = H + (size_t)rowbase * U_DIM;
    #pragma unroll
    for (int j = 0; j < 2; ++j) {
      const int e = (tid * 2 + j) * 8;
      h_async_b128(src + e, &h_tile[0][e]);
    }
    h_async_wait();
  }
  __syncthreads();

  for (int t = 0; t < T_LEN; ++t) {
    const int cur = t & 1;

    // Kick off DMA of next step's H tile; overlaps the WMMA phase below.
    if (t + 1 < T_LEN) {
      const __bf16* src = H + ((size_t)(t + 1) * B_SZ + rowbase) * U_DIM;
      #pragma unroll
      for (int j = 0; j < 2; ++j) {
        const int e = (tid * 2 + j) * 8;
        h_async_b128(src + e, &h_tile[cur ^ 1][e]);
      }
    }

    // S = y_prev @ W2 (two 16-col slices): 16 LDS A-loads, 32 WMMAs, no VMEM.
    v8f acc0 = {}, acc1 = {};
    #pragma unroll
    for (int kc = 0; kc < 16; ++kc) {
      v16bf a = *(const v16bf*)(&y_lds[(lane & 15) * U_DIM + kc * 32 + kh * 16]);
      acc0 = __builtin_amdgcn_wmma_f32_16x16x32_bf16(
          false, a, false, Bf0[kc], (short)0, acc0, false, false);
      acc1 = __builtin_amdgcn_wmma_f32_16x16x32_bf16(
          false, a, false, Bf1[kc], (short)0, acc1, false, false);
    }
    __syncthreads();  // all reads of y_{t-1} complete before overwrite

    // Epilogue: y = H_t + tanh(S + b2); write state; partial dot with Wc.
    float part[8];
    #pragma unroll
    for (int r = 0; r < 8; ++r) {
      const int m = r + 8 * kh;  // local row 0..15
      const float v0 = (float)h_tile[cur][m * U_DIM + n0] + tanhf(acc0[r] + bias0);
      const float v1 = (float)h_tile[cur][m * U_DIM + n1] + tanhf(acc1[r] + bias1);
      y_lds[m * U_DIM + n0] = (__bf16)v0;
      y_lds[m * U_DIM + n1] = (__bf16)v1;
      part[r] = v0 * wc0 + v1 * wc1;
    }

    // Reduce dot partials over the 16 columns held by each half-wave
    // (xor masks 1..8 stay within a 16-lane half), then 256 LDS atomics/step.
    #pragma unroll
    for (int r = 0; r < 8; ++r) {
      #pragma unroll
      for (int s = 1; s < 16; s <<= 1)
        part[r] += __shfl_xor(part[r], s, 32);
    }
    if (nloc == 0) {
      #pragma unroll
      for (int r = 0; r < 8; ++r)
        atomicAdd(&accum[r + 8 * kh], part[r]);
    }

    h_async_wait();   // this wave's portion of the t+1 tile has landed
    __syncthreads();  // all portions landed; atomics visible

    if (tid < 16) {
      out[(size_t)(rowbase + tid) * T_LEN + t] = accum[tid] + bcv;
      accum[tid] = 0.0f;
    }
    __syncthreads();
  }
}

// ---------------------------------------------------------------------------
extern "C" void kernel_launch(void* const* d_in, const int* in_sizes, int n_in,
                              void* d_out, int out_size, void* d_ws, size_t ws_size,
                              hipStream_t stream) {
  (void)in_sizes; (void)n_in; (void)out_size; (void)ws_size;
  const float* x  = (const float*)d_in[0];
  const float* W1 = (const float*)d_in[1];
  const float* b1 = (const float*)d_in[2];
  const float* W2 = (const float*)d_in[3];
  const float* b2 = (const float*)d_in[4];
  const float* Wc = (const float*)d_in[5];
  const float* bc = (const float*)d_in[6];
  float* out = (float*)d_out;

  // Workspace layout: W1t (512KB) | W2t (512KB) | H (256MB bf16, time-major)
  char* ws = (char*)d_ws;
  __bf16* W1t = (__bf16*)(ws);
  __bf16* W2t = (__bf16*)(ws + (size_t)U_DIM * D_DIM * 2);
  __bf16* H   = (__bf16*)(ws + (size_t)2 * U_DIM * D_DIM * 2);

  prep_weights_kernel<<<(U_DIM * D_DIM) / 256, 256, 0, stream>>>(W1, W2, W1t, W2t);
  input_gemm_tanh_kernel<<<(B_SZ * T_LEN) / 64, 512, 0, stream>>>(x, W1t, b1, H);
  recurrence_kernel<<<B_SZ / 16, 512, 0, stream>>>(H, W2t, b2, Wc, bc, out);
}
